// VectorQuantizer_89111981457821
// MI455X (gfx1250) — matrix operand order repaired and checked by reading, not compiled, not run on hardware
//
#include <hip/hip_runtime.h>
#include <hip/hip_bf16.h>

#define DIM    256
#define NEMB   8192
#define NROWS  16384
#define NTILE  (NEMB / 16)     // 512 column tiles
#define DECAYF 0.99f
#define OMDEC  0.01f
#define EPSF   1e-5f

typedef __attribute__((ext_vector_type(16))) __bf16 v16bf;
typedef __attribute__((ext_vector_type(8)))  __bf16 v8bf;
typedef __attribute__((ext_vector_type(8)))  float  v8f;

union AFrag { v16bf v; v8bf h[2]; };

#define WMMA_BF16(a, b, c) \
    __builtin_amdgcn_wmma_f32_16x16x32_bf16(false, (a), false, (b), (short)0, (c), false, false)

// ---------------------------------------------------------------------------
// Prep: split z (f32) into bf16 hi/lo, elementwise.
// ---------------------------------------------------------------------------
__global__ void prep_x_kernel(const float* __restrict__ z,
                              __bf16* __restrict__ xhi,
                              __bf16* __restrict__ xlo) {
    int i = blockIdx.x * 256 + threadIdx.x;   // exactly NROWS*DIM threads
    float v = z[i];
    __bf16 h = (__bf16)v;
    xhi[i] = h;
    xlo[i] = (__bf16)(v - (float)h);
}

// ---------------------------------------------------------------------------
// Prep: transpose embedding [DIM, NEMB] -> Et [NEMB, DIM], split bf16 hi/lo.
// LDS-tiled 32x32 transpose, coalesced on both sides.
// ---------------------------------------------------------------------------
__global__ void prep_embed_kernel(const float* __restrict__ E,
                                  __bf16* __restrict__ ethi,
                                  __bf16* __restrict__ etlo) {
    __shared__ float tile[32][33];
    int jt = blockIdx.x * 32;       // code (n_embed) tile
    int dt = blockIdx.y * 32;       // dim tile
    int tx = threadIdx.x;           // 0..31
    int ty = threadIdx.y;           // 0..7
    for (int i = ty; i < 32; i += 8)
        tile[i][tx] = E[(dt + i) * NEMB + jt + tx];   // coalesced in tx (j)
    __syncthreads();
    for (int i = ty; i < 32; i += 8) {
        float v = tile[tx][i];                        // d = tx, j = i
        __bf16 h = (__bf16)v;
        int o = (jt + i) * DIM + dt + tx;             // coalesced in tx (d)
        ethi[o] = h;
        etlo[o] = (__bf16)(v - (float)h);
    }
}

// ---------------------------------------------------------------------------
// Prep: e2[j] = sum_d embedding[d][j]^2, coalesced (thread = column j).
// ---------------------------------------------------------------------------
__global__ void e2_kernel(const float* __restrict__ E, float* __restrict__ e2) {
    int j = blockIdx.x * 256 + threadIdx.x;
    float s = 0.f;
    for (int d = 0; d < DIM; ++d) {
        float v = E[d * NEMB + j];
        s += v * v;
    }
    e2[j] = s;
}

// ---------------------------------------------------------------------------
// Zero workspace accumulators.
// ---------------------------------------------------------------------------
__global__ void zero_kernel(float* __restrict__ p, int n) {
    int i = blockIdx.x * 256 + threadIdx.x;
    if (i < n) p[i] = 0.f;
}

// ---------------------------------------------------------------------------
// Core: fused GEMM + argmin via bf16 WMMA with hi/lo error compensation.
// dist[m][n] = e2[n] - 2*dot(x_m, e_n)  (||x||^2 is row-constant -> dropped).
//
// 8 waves/block, 16 rows/wave -> 128 rows/block.  The 16-code B tile (hi+lo,
// 16 KB) is staged into LDS once per tile and shared by all 8 waves, double-
// buffered so the next tile's global loads overlap the current WMMA chain.
// Three independent accumulator chains break the 24-deep WMMA RAW chain.
// ---------------------------------------------------------------------------
__global__ __launch_bounds__(256, 1)
void argmin_wmma_kernel(const __bf16* __restrict__ xhi,
                        const __bf16* __restrict__ xlo,
                        const __bf16* __restrict__ ethi,
                        const __bf16* __restrict__ etlo,
                        const float*  __restrict__ e2,
                        int* __restrict__ embed_ind) {
    // [buf][ hi rows 16x256 | lo rows 16x256 ] elements, 16 KB per buffer
    __shared__ __align__(16) __bf16 sB[2][8192];

    const int tid   = threadIdx.x;
    const int wave  = tid >> 5;
    const int lane  = tid & 31;
    const int l16   = lane & 15;
    const int lhalf = lane >> 4;
    const int m0    = blockIdx.x * 128 + wave * 16;

    // Preload A fragments (hi and lo) for this wave's 16 rows, full K=256.
    // A k order chosen here (same as B): lane half h holds contiguous K in [32k+16h, +16).
    v16bf aHi[8], aLo[8];
    {
        const __bf16* ah = xhi + (size_t)(m0 + l16) * DIM + 16 * lhalf;
        const __bf16* al = xlo + (size_t)(m0 + l16) * DIM + 16 * lhalf;
#pragma unroll
        for (int k = 0; k < 8; ++k) {
            AFrag fh, fl;
            fh.h[0] = *(const v8bf*)(ah + 32 * k);
            fh.h[1] = *(const v8bf*)(ah + 32 * k + 8);
            fl.h[0] = *(const v8bf*)(al + 32 * k);
            fl.h[1] = *(const v8bf*)(al + 32 * k + 8);
            aHi[k] = fh.v;
            aLo[k] = fl.v;
        }
    }

    float minv[8];
    int   mini[8];
#pragma unroll
    for (int r = 0; r < 8; ++r) { minv[r] = 3.4e38f; mini[r] = 0; }

    // Prologue: stage tile 0 into buffer 0 (each tile region is 8 KB contiguous).
    {
        const v8bf* srcH = (const v8bf*)ethi;
        const v8bf* srcL = (const v8bf*)etlo;
        v8bf* dst = (v8bf*)(&sB[0][0]);
        dst[tid]             = srcH[tid];
        dst[tid + 256]       = srcH[tid + 256];
        dst[tid + 512]       = srcL[tid];
        dst[tid + 768]       = srcL[tid + 256];
    }
    __syncthreads();

    for (int t = 0; t < NTILE; ++t) {
        const int buf = t & 1;

        // Prefetch the tile after next into cache (global_prefetch_b8).
        if (t + 2 < NTILE) {
            __builtin_prefetch((const char*)(ethi + (size_t)(t + 2) * 4096) + tid * 64, 0, 0);
            __builtin_prefetch((const char*)(etlo + (size_t)(t + 2) * 4096) + tid * 64, 0, 0);
        }

        // Issue next tile's global loads early; consumed after the WMMA phase.
        v8bf h0, h1, l0, l1;
        const bool havenext = (t + 1 < NTILE);
        if (havenext) {
            const v8bf* srcH = (const v8bf*)(ethi + (size_t)(t + 1) * 4096);
            const v8bf* srcL = (const v8bf*)(etlo + (size_t)(t + 1) * 4096);
            h0 = srcH[tid];
            h1 = srcH[tid + 256];
            l0 = srcL[tid];
            l1 = srcL[tid + 256];
        }

        // WMMA phase: B fragments from LDS.
        // B k order chosen here (same as A): lane half h holds contiguous K in [32k+16h, +16).
        const __bf16* bh = &sB[buf][l16 * 256 + 16 * lhalf];
        const __bf16* bl = bh + 4096;
        v8f acc0 = {}, acc1 = {}, acc2 = {};
#pragma unroll
        for (int k = 0; k < 8; ++k) {
            v16bf bHi = *(const v16bf*)(bh + 32 * k);
            v16bf bLo = *(const v16bf*)(bl + 32 * k);
            acc0 = WMMA_BF16(aHi[k], bHi, acc0);
            acc1 = WMMA_BF16(aLo[k], bHi, acc1);
            acc2 = WMMA_BF16(aHi[k], bLo, acc2);
        }

        // Commit staged data into the other buffer.
        if (havenext) {
            v8bf* dst = (v8bf*)(&sB[buf ^ 1][0]);
            dst[tid]       = h0;
            dst[tid + 256] = h1;
            dst[tid + 512] = l0;
            dst[tid + 768] = l1;
        }

        // Argmin update.  C/D layout: acc[r] is (M = r + 8*lhalf, N = t*16 + l16).
        const int   n   = t * 16 + l16;
        const float en2 = e2[n];
#pragma unroll
        for (int r = 0; r < 8; ++r) {
            float dv = en2 - 2.0f * (acc0[r] + acc1[r] + acc2[r]);
            if (dv < minv[r]) { minv[r] = dv; mini[r] = n; }
        }

        __syncthreads();
    }

    // Reduce (min, argmin) across the 16 lanes holding the same rows.
#pragma unroll
    for (int off = 1; off < 16; off <<= 1) {
#pragma unroll
        for (int r = 0; r < 8; ++r) {
            float ov = __shfl_xor(minv[r], off, 32);
            int   oi = __shfl_xor(mini[r], off, 32);
            if (ov < minv[r] || (ov == minv[r] && oi < mini[r])) {
                minv[r] = ov; mini[r] = oi;
            }
        }
    }
    if (l16 == 0) {
#pragma unroll
        for (int r = 0; r < 8; ++r)
            embed_ind[m0 + lhalf * 8 + r] = mini[r];
    }
}

// ---------------------------------------------------------------------------
// Gather quantize (== quantize_st forward value), diff sum, index output.
// ---------------------------------------------------------------------------
__global__ void quantize_kernel(const float* __restrict__ z,
                                const int*   __restrict__ ind,
                                const float* __restrict__ E,
                                float* __restrict__ q_out,
                                float* __restrict__ ind_out,
                                float* __restrict__ diff_acc) {
    int n = blockIdx.x;
    int d = threadIdx.x;
    int idx = ind[n];
    float q  = E[(size_t)d * NEMB + idx];   // L2-resident gather
    float zv = z[(size_t)n * DIM + d];
    q_out[(size_t)n * DIM + d] = q;         // z + sg(q - z) == q in value
    float e = q - zv;
    __shared__ float red[256];
    red[d] = e * e;
    __syncthreads();
    for (int s = 128; s > 0; s >>= 1) {
        if (d < s) red[d] += red[d + s];
        __syncthreads();
    }
    if (d == 0) {
        atomicAdd(diff_acc, red[0]);
        ind_out[n] = (float)idx;
    }
}

// ---------------------------------------------------------------------------
// Segment sums: counts and per-code feature sums (transposed accumulator).
// ---------------------------------------------------------------------------
__global__ void scatter_kernel(const float* __restrict__ z,
                               const int*   __restrict__ ind,
                               float* __restrict__ countT,
                               float* __restrict__ sumT) {
    int n = blockIdx.x;
    int d = threadIdx.x;
    int idx = ind[n];
    if (d == 0) atomicAdd(&countT[idx], 1.0f);
    atomicAdd(&sumT[(size_t)idx * DIM + d], z[(size_t)n * DIM + d]);
}

// ---------------------------------------------------------------------------
// Finalize 1: new_cluster_size + its total sum n.
// ---------------------------------------------------------------------------
__global__ void finalize_cs_kernel(const float* __restrict__ cs,
                                   const float* __restrict__ countT,
                                   float* __restrict__ ncs_out,
                                   float* __restrict__ n_acc) {
    int j = blockIdx.x * 256 + threadIdx.x;
    float v = cs[j] * DECAYF + OMDEC * countT[j];
    ncs_out[j] = v;
    __shared__ float red[256];
    red[threadIdx.x] = v;
    __syncthreads();
    for (int s = 128; s > 0; s >>= 1) {
        if (threadIdx.x < s) red[threadIdx.x] += red[threadIdx.x + s];
        __syncthreads();
    }
    if (threadIdx.x == 0) atomicAdd(n_acc, red[0]);
}

// ---------------------------------------------------------------------------
// Finalize 2: new_embedding_avg and new_embedding.
// ---------------------------------------------------------------------------
__global__ void finalize_emb_kernel(const float* __restrict__ eavg,
                                    const float* __restrict__ sumT,
                                    const float* __restrict__ ncs,
                                    const float* __restrict__ n_acc,
                                    float* __restrict__ navg_out,
                                    float* __restrict__ nemb_out) {
    int d = blockIdx.y;
    int j = blockIdx.x * 256 + threadIdx.x;
    float n  = *n_acc;
    float cs = ncs[j];
    float smoothed = (cs + EPSF) / (n + (float)NEMB * EPSF) * n;
    float na = eavg[(size_t)d * NEMB + j] * DECAYF +
               OMDEC * sumT[(size_t)j * DIM + d];
    navg_out[(size_t)d * NEMB + j] = na;
    nemb_out[(size_t)d * NEMB + j] = na / smoothed;
}

__global__ void finalize_diff_kernel(const float* __restrict__ diff_acc,
                                     float* __restrict__ diff_out) {
    diff_out[0] = diff_acc[0] * (1.0f / ((float)NROWS * (float)DIM));
}

// ---------------------------------------------------------------------------
// Host-side launch.
// ---------------------------------------------------------------------------
extern "C" void kernel_launch(void* const* d_in, const int* in_sizes, int n_in,
                              void* d_out, int out_size, void* d_ws, size_t ws_size,
                              hipStream_t stream) {
    const float* z    = (const float*)d_in[0];   // [16,32,32,256]
    const float* E    = (const float*)d_in[1];   // [256, 8192]
    const float* cs   = (const float*)d_in[2];   // [8192]
    const float* eavg = (const float*)d_in[3];   // [256, 8192]

    // ---- workspace layout (bytes) ----
    char* ws = (char*)d_ws;
    __bf16* xhi   = (__bf16*)(ws + 0);                         // 8 MB
    __bf16* xlo   = (__bf16*)(ws + 8388608);                   // 8 MB
    __bf16* ethi  = (__bf16*)(ws + 16777216);                  // 4 MB
    __bf16* etlo  = (__bf16*)(ws + 20971520);                  // 4 MB
    float*  e2    = (float*) (ws + 25165824);                  // 32 KB
    int*    ind   = (int*)   (ws + 25198592);                  // 64 KB
    float*  countT= (float*) (ws + 25264128);                  // 32 KB
    float*  sumT  = (float*) (ws + 25296896);                  // 8 MB
    float*  diff_acc = (float*)(ws + 33685504);                // 4 B
    float*  n_acc    = (float*)(ws + 33685508);                // 4 B

    // ---- output layout (floats, reference return order) ----
    float* out       = (float*)d_out;
    float* q_out     = out;                       // 4194304
    float* diff_out  = out + 4194304;             // 1
    float* ind_out   = out + 4194305;             // 16384
    float* nemb_out  = out + 4210689;             // 2097152
    float* ncs_out   = out + 6307841;             // 8192
    float* navg_out  = out + 6316033;             // 2097152

    // Prep
    prep_x_kernel<<<NROWS * DIM / 256, 256, 0, stream>>>(z, xhi, xlo);
    prep_embed_kernel<<<dim3(NEMB / 32, DIM / 32), dim3(32, 8), 0, stream>>>(E, ethi, etlo);
    e2_kernel<<<NEMB / 256, 256, 0, stream>>>(E, e2);

    // Zero accumulators (countT..n_acc is contiguous)
    const int nz = NEMB + NEMB * DIM + 2;
    zero_kernel<<<(nz + 255) / 256, 256, 0, stream>>>(countT, nz);

    // Fused GEMM + argmin (WMMA), 128 rows per 256-thread block
    argmin_wmma_kernel<<<NROWS / 128, 256, 0, stream>>>(xhi, xlo, ethi, etlo, e2, ind);

    // Gather + diff, segment sums
    quantize_kernel<<<NROWS, 256, 0, stream>>>(z, ind, E, q_out, ind_out, diff_acc);
    scatter_kernel<<<NROWS, 256, 0, stream>>>(z, ind, countT, sumT);

    // EMA finalize
    finalize_cs_kernel<<<NEMB / 256, 256, 0, stream>>>(cs, countT, ncs_out, n_acc);
    finalize_emb_kernel<<<dim3(NEMB / 256, DIM), 256, 0, stream>>>(
        eavg, sumT, ncs_out, n_acc, navg_out, nemb_out);
    finalize_diff_kernel<<<1, 1, 0, stream>>>(diff_acc, diff_out);
}
